// MHA_4054449127906
// MI455X (gfx1250) — matrix operand order repaired and checked by reading, not compiled, not run on hardware
//
#include <hip/hip_runtime.h>
#include <hip/hip_bf16.h>

typedef __attribute__((ext_vector_type(16))) __bf16 v16bf;
typedef __attribute__((ext_vector_type(8)))  float  v8f;
typedef __attribute__((ext_vector_type(4)))  unsigned int u32x4;
typedef __attribute__((ext_vector_type(8)))  unsigned int u32x8;

union Frag16 { v16bf v; unsigned short u[16]; };

constexpr int BB  = 4;
constexpr int SS  = 2048;
constexpr int HH  = 2048;
constexpr int NHD = 16;
constexpr int HDD = 128;

__device__ __forceinline__ unsigned short f2bf(float x) {
  unsigned int u = __float_as_uint(x);
  u += 0x7fffu + ((u >> 16) & 1u);   // round-to-nearest-even
  return (unsigned short)(u >> 16);
}

__device__ __forceinline__ v8f wmma_bf16(v16bf a, v16bf b, v8f c) {
  // D = A(16x32) * B(32x16) + C, f32 accumulate
  return __builtin_amdgcn_wmma_f32_16x16x32_bf16(false, a, false, b, (short)0, c,
                                                 false, false);
}

// A-fragment (16x32 bf16) from LDS row-major tile [row][k]:
// M=lane%16; VGPR g: K=2g (+16 for lanes16-31), matching the B fragments
__device__ __forceinline__ v16bf a_frag_lds(const unsigned short* t, int stride,
                                            int rowBase) {
  const int lane = threadIdx.x & 31;
  const int r = rowBase + (lane & 15);
  const int h = lane >> 4;
  Frag16 f;
#pragma unroll
  for (int g = 0; g < 8; ++g) {
    int kb = 2 * g + 16 * h;
    f.u[2 * g]     = t[r * stride + kb];
    f.u[2 * g + 1] = t[r * stride + kb + 1];
  }
  return f.v;
}

// A-fragment from global bf16 row-major (stride elems), K window [kBase, kBase+32)
__device__ __forceinline__ v16bf a_frag_global(const unsigned short* base,
                                               int stride, int kBase) {
  const int lane = threadIdx.x & 31;
  const int r = lane & 15;
  const int h = lane >> 4;
  Frag16 f;
#pragma unroll
  for (int g = 0; g < 8; ++g) {
    int kb = kBase + 2 * g + 16 * h;
    unsigned int pair = *(const unsigned int*)(base + (size_t)r * stride + kb);
    f.u[2 * g]     = (unsigned short)(pair & 0xffffu);
    f.u[2 * g + 1] = (unsigned short)(pair >> 16);
  }
  return f.v;
}

// B-fragment (32x16 bf16) from LDS tile stored [K][N] row-major (column gather)
__device__ __forceinline__ v16bf b_frag_lds(const unsigned short* t, int stride,
                                            int colBase) {
  const int lane = threadIdx.x & 31;
  const int n = colBase + (lane & 15);
  const int h = lane >> 4;
  Frag16 f;
#pragma unroll
  for (int g = 0; g < 8; ++g) {
    int k = 2 * g + 16 * h;
    f.u[2 * g]     = t[k * stride + n];
    f.u[2 * g + 1] = t[(k + 1) * stride + n];
  }
  return f.v;
}

// B-fragment for B = T^T where T is stored [row][feature] row-major in LDS:
// element(kk, n) = T[rowBase+n][kBase+kk]  -> contiguous pairs, b128-mergeable
__device__ __forceinline__ v16bf bT_frag_lds(const unsigned short* t, int stride,
                                             int rowBase, int kBase) {
  const int lane = threadIdx.x & 31;
  const int n = lane & 15;
  const int h = lane >> 4;
  const unsigned short* rp = t + (rowBase + n) * stride + kBase;
  Frag16 f;
#pragma unroll
  for (int g = 0; g < 8; ++g) {
    int kk = 2 * g + 16 * h;
    unsigned int pair = *(const unsigned int*)(rp + kk);
    f.u[2 * g]     = (unsigned short)(pair & 0xffffu);
    f.u[2 * g + 1] = (unsigned short)(pair >> 16);
  }
  return f.v;
}

// ---------------------------------------------------------------------------
// Tensor Data Mover: 2D tile (bf16) global -> LDS, with LDS padding applied by
// the TDM itself (D# pad_interval/pad_amount). Issued once per wave; tracked
// by TENSORcnt. D# layout per CDNA5 ISA ch.8 (groups 0 and 1; 2D tensor).
// ---------------------------------------------------------------------------
__device__ __forceinline__ void tdm_load_2d(unsigned lds_addr, const void* gptr,
                                            unsigned tensor_d0, unsigned tensor_d1,
                                            unsigned long long stride0,
                                            unsigned tile_d0, unsigned tile_d1,
                                            unsigned pad_interval,
                                            unsigned pad_amount) {
  union G0 { unsigned long long q[2]; u32x4 v; } g0;
  union G1 { unsigned long long q[4]; u32x8 v; } g1;
  unsigned long long ga = (unsigned long long)gptr;
  g0.q[0] = 1ull | ((unsigned long long)lds_addr << 32);        // count=1 | lds_addr
  g0.q[1] = (ga & 0x01FFFFFFFFFFFFFFull) | (2ull << 62);        // global_addr | type=2
  g1.q[0] = (1ull << 16)                                        // data_size = 2 bytes
          | (1ull << 20)                                        // pad_enable
          | ((unsigned long long)pad_interval << 22)
          | ((unsigned long long)pad_amount << 25)
          | (((unsigned long long)(tensor_d0 & 0xffffu)) << 48);// tensor_dim0[15:0]
  g1.q[1] = ((unsigned long long)tensor_d0 >> 16)               // tensor_dim0[31:16]
          | ((unsigned long long)tensor_d1 << 16)               // tensor_dim1
          | ((unsigned long long)tile_d0 << 48);                // tile_dim0
  g1.q[2] = (unsigned long long)tile_d1                         // tile_dim1
          | ((stride0 & 0xffffffffull) << 32);                  // dim0_stride[31:0]
  g1.q[3] = (stride0 >> 32) & 0xffffull;                        // dim0_stride[47:32]
  asm volatile("tensor_load_to_lds %0, %1" :: "s"(g0.v), "s"(g1.v) : "memory");
}

// ---------------------------------------------------------------------------
// GEMM: C[M=BB*SS, N=HH] = A[M,K=HH] (fp32 or bf16) x W[K,N] (fp32) + bias
// OUT_MODE 0: fp32 row-major.  OUT_MODE 1: bf16 -> [B,nh,S,hd].
// OUT_MODE 2: bf16 -> [B,nh,hd,S]  (transposed, for V).
// Block = 256 threads (8 waves, 2x4), tile 128x128, K-step 32.
// ---------------------------------------------------------------------------
template <bool A_BF16, int OUT_MODE>
__global__ __launch_bounds__(256) void gemm_bias_k(
    const void* __restrict__ Ap, const float* __restrict__ W,
    const float* __restrict__ bias, void* __restrict__ outp) {
  constexpr int N = HH, K = HH;
  __shared__ unsigned short sA[128 * 40];   // padded stride: conflict-free A reads
  __shared__ unsigned short sB[32 * 136];
  const int tid  = threadIdx.x;
  const int lane = tid & 31;
  const int w    = tid >> 5;
  const int wm   = w >> 2;                  // 0..1 -> 64 rows each
  const int wn   = w & 3;                   // 0..3 -> 32 cols each
  const int mbase = blockIdx.y * 128;
  const int nbase = blockIdx.x * 128;
  v8f acc[4][2] = {};

  for (int k0 = 0; k0 < K; k0 += 32) {
    if constexpr (A_BF16) {
      const unsigned short* A = (const unsigned short*)Ap;
#pragma unroll
      for (int i = 0; i < 2; ++i) {
        int idx = tid + i * 256;            // 512 x uint4 (8 bf16)
        int row = idx >> 2, c8 = idx & 3;
        uint4 val = *(const uint4*)(A + (size_t)(mbase + row) * K + k0 + c8 * 8);
        *(uint4*)(sA + row * 40 + c8 * 8) = val;
      }
      if (k0 + 32 < K)
        __builtin_prefetch(A + (size_t)(mbase + (tid >> 1)) * K + k0 + 32 +
                               (tid & 1) * 16, 0, 1);
    } else {
      const float* A = (const float*)Ap;
#pragma unroll
      for (int i = 0; i < 4; ++i) {
        int idx = tid + i * 256;            // 1024 x float4
        int row = idx >> 3, c4 = idx & 7;
        float4 v4 = *(const float4*)(A + (size_t)(mbase + row) * K + k0 + c4 * 4);
        unsigned int p0 = (unsigned int)f2bf(v4.x) | ((unsigned int)f2bf(v4.y) << 16);
        unsigned int p1 = (unsigned int)f2bf(v4.z) | ((unsigned int)f2bf(v4.w) << 16);
        unsigned int* dst = (unsigned int*)(sA + row * 40 + c4 * 4);
        dst[0] = p0; dst[1] = p1;
      }
      if (k0 + 32 < K)
        __builtin_prefetch(A + (size_t)(mbase + (tid >> 1)) * K + k0 + 32 +
                               (tid & 1) * 16, 0, 1);
    }
#pragma unroll
    for (int i = 0; i < 4; ++i) {
      int idx = tid + i * 256;              // 1024 x float4 of W tile 32x128
      int row = idx >> 5, c4 = idx & 31;
      float4 v4 = *(const float4*)(W + (size_t)(k0 + row) * N + nbase + c4 * 4);
      unsigned int p0 = (unsigned int)f2bf(v4.x) | ((unsigned int)f2bf(v4.y) << 16);
      unsigned int p1 = (unsigned int)f2bf(v4.z) | ((unsigned int)f2bf(v4.w) << 16);
      unsigned int* dst = (unsigned int*)(sB + row * 136 + c4 * 4);
      dst[0] = p0; dst[1] = p1;
    }
    if (k0 + 32 < K)
      __builtin_prefetch(W + (size_t)(k0 + 32 + (tid >> 3)) * N + nbase +
                             (tid & 7) * 16, 0, 1);
    __syncthreads();

    v16bf aF[4], bF[2];
#pragma unroll
    for (int mt = 0; mt < 4; ++mt) aF[mt] = a_frag_lds(sA, 40, wm * 64 + mt * 16);
#pragma unroll
    for (int nt = 0; nt < 2; ++nt) bF[nt] = b_frag_lds(sB, 136, wn * 32 + nt * 16);
#pragma unroll
    for (int mt = 0; mt < 4; ++mt)
#pragma unroll
      for (int nt = 0; nt < 2; ++nt)
        acc[mt][nt] = wmma_bf16(aF[mt], bF[nt], acc[mt][nt]);
    __syncthreads();
  }

  const int n = lane & 15, h = lane >> 4;
#pragma unroll
  for (int mt = 0; mt < 4; ++mt)
#pragma unroll
    for (int nt = 0; nt < 2; ++nt)
#pragma unroll
      for (int g = 0; g < 8; ++g) {
        int row = mbase + wm * 64 + mt * 16 + g + 8 * h;
        int col = nbase + wn * 32 + nt * 16 + n;
        float val = acc[mt][nt][g] + bias[col];
        if constexpr (OUT_MODE == 0) {
          ((float*)outp)[(size_t)row * N + col] = val;
        } else if constexpr (OUT_MODE == 1) {
          int b = row >> 11, s = row & (SS - 1);
          int head = col >> 7, d = col & (HDD - 1);
          ((unsigned short*)outp)[(((size_t)(b * NHD + head)) * SS + s) * HDD + d] =
              f2bf(val);
        } else {  // OUT_MODE == 2: V transposed [B,nh,hd,S]
          int b = row >> 11, s = row & (SS - 1);
          int head = col >> 7, d = col & (HDD - 1);
          ((unsigned short*)outp)[(((size_t)(b * NHD + head)) * HDD + d) * SS + s] =
              f2bf(val);
        }
      }
}

// ---------------------------------------------------------------------------
// Flash attention: grid (S/128, B*nh); block = 256 threads = 8 waves.
// Each wave owns 16 q-rows; Q in registers; K [key][d] and V^T [d][key] tiles
// streamed via double-buffered TDM (wave 0 issues, TENSORcnt-tracked, TDM
// applies the LDS bank padding); streaming softmax; P through per-wave LDS.
// ---------------------------------------------------------------------------
__global__ __launch_bounds__(256) void flash_attn_k(
    const unsigned short* __restrict__ Q, const unsigned short* __restrict__ Kt,
    const unsigned short* __restrict__ Vt, unsigned short* __restrict__ Out) {
  __shared__ unsigned short sK[2][32 * 136];   // [key][d], padded by TDM
  __shared__ unsigned short sV[2][128 * 40];   // [d][key], padded by TDM
  __shared__ unsigned short sP[8][16 * 34];    // per-wave P strip (16x32)
  const int tid = threadIdx.x, lane = tid & 31, w = tid >> 5;
  const int bh = blockIdx.y;                   // b * nh + head
  const int qbase = blockIdx.x * 128 + w * 16;
  const int n = lane & 15, h = lane >> 4;
  const size_t headoff = (size_t)bh * SS * HDD;
  const unsigned short* Kbase = Kt + headoff;  // [key][d]
  const unsigned short* Vbase = Vt + headoff;  // [d][key]

  v16bf qf[4];
#pragma unroll
  for (int c = 0; c < 4; ++c)
    qf[c] = a_frag_global(Q + headoff + (size_t)qbase * HDD, HDD, c * 32);

  v8f o[8] = {};
  float m[8], l[8];
#pragma unroll
  for (int g = 0; g < 8; ++g) { m[g] = -1e30f; l[g] = 0.0f; }
  const float scale = 0.08838834764831843f;    // 1/sqrt(128)

  auto issue_tiles = [&](int j0, int buf) {
    // K tile: 32 rows x 128 d; row = 256B = 64 DW -> interval 5, pad 4 DW -> 136
    tdm_load_2d((unsigned)(size_t)&sK[buf][0], Kbase + (size_t)j0 * HDD,
                HDD, SS, HDD, HDD, 32, 5, 3);
    // V^T tile: 128 rows x 32 keys; row = 64B = 16 DW -> interval 3, pad 4 DW -> 40
    tdm_load_2d((unsigned)(size_t)&sV[buf][0], Vbase + j0,
                SS, HDD, SS, 32, HDD, 3, 3);
  };

  if (w == 0) issue_tiles(0, 0);

  int buf = 0;
  for (int j0 = 0; j0 < SS; j0 += 32, buf ^= 1) {
    if (w == 0) {
      if (j0 + 32 < SS) {
        issue_tiles(j0 + 32, buf ^ 1);               // prefetch next stage
        __builtin_amdgcn_s_wait_tensorcnt(2);        // current stage complete
      } else {
        __builtin_amdgcn_s_wait_tensorcnt(0);
      }
    }
    __syncthreads();                                 // publish tile to all waves

    // S = Q * K^T over hd=128 (4 chained WMMAs per 16-key subtile)
    v8f sc[2] = {};
#pragma unroll
    for (int jt = 0; jt < 2; ++jt)
#pragma unroll
      for (int c = 0; c < 4; ++c)
        sc[jt] = wmma_bf16(qf[c], bT_frag_lds(&sK[buf][0], 136, jt * 16, c * 32),
                           sc[jt]);

    // streaming softmax (row stats reduced across the 16-lane column groups)
    float rmax[8], rsum[8];
#pragma unroll
    for (int g = 0; g < 8; ++g) {
      sc[0][g] *= scale; sc[1][g] *= scale;
      rmax[g] = fmaxf(sc[0][g], sc[1][g]);
    }
#pragma unroll
    for (int mk = 8; mk >= 1; mk >>= 1)
#pragma unroll
      for (int g = 0; g < 8; ++g)
        rmax[g] = fmaxf(rmax[g], __shfl_xor(rmax[g], mk, 32));
#pragma unroll
    for (int g = 0; g < 8; ++g) {
      float mn = fmaxf(m[g], rmax[g]);
      float alpha = __expf(m[g] - mn);
      float p0 = __expf(sc[0][g] - mn);
      float p1 = __expf(sc[1][g] - mn);
      sc[0][g] = p0; sc[1][g] = p1;
      rsum[g] = p0 + p1;
      m[g] = mn;
      l[g] *= alpha;
#pragma unroll
      for (int nt = 0; nt < 8; ++nt) o[nt][g] *= alpha;
    }
#pragma unroll
    for (int mk = 8; mk >= 1; mk >>= 1)
#pragma unroll
      for (int g = 0; g < 8; ++g)
        rsum[g] += __shfl_xor(rsum[g], mk, 32);
#pragma unroll
    for (int g = 0; g < 8; ++g) l[g] += rsum[g];

    // P (D-layout) -> wave-private LDS -> A-fragment (same-wave LDS in-order)
    unsigned short* pw = sP[w];
#pragma unroll
    for (int jt = 0; jt < 2; ++jt)
#pragma unroll
      for (int g = 0; g < 8; ++g)
        pw[(g + 8 * h) * 34 + jt * 16 + n] = f2bf(sc[jt][g]);

    v16bf pf = a_frag_lds(pw, 34, 0);
#pragma unroll
    for (int nt = 0; nt < 8; ++nt)
      o[nt] = wmma_bf16(pf, bT_frag_lds(&sV[buf][0], 40, nt * 16, 0), o[nt]);
    __syncthreads();                                 // retire tile buffer
  }

  const int b = bh >> 4, hh = bh & 15;               // bh / nh, bh % nh
#pragma unroll
  for (int g = 0; g < 8; ++g) {
    float inv = 1.0f / l[g];
    int s = qbase + g + 8 * h;
#pragma unroll
    for (int nt = 0; nt < 8; ++nt)
      Out[((size_t)b * SS + s) * HH + hh * HDD + nt * 16 + n] =
          f2bf(o[nt][g] * inv);
  }
}

// ---------------------------------------------------------------------------
extern "C" void kernel_launch(void* const* d_in, const int* in_sizes, int n_in,
                              void* d_out, int out_size, void* d_ws, size_t ws_size,
                              hipStream_t stream) {
  const float* x  = (const float*)d_in[0];
  const float* Wq = (const float*)d_in[1];
  const float* bq = (const float*)d_in[2];
  const float* Wk = (const float*)d_in[3];
  const float* bk = (const float*)d_in[4];
  const float* Wv = (const float*)d_in[5];
  const float* bv = (const float*)d_in[6];
  const float* Wo = (const float*)d_in[7];
  const float* bo = (const float*)d_in[8];

  // ws layout (bf16): Q | K | V^T | attn-out, 32 MB each = 128 MB total
  const size_t elems = (size_t)BB * SS * HH;
  unsigned short* qw = (unsigned short*)d_ws;
  unsigned short* kw = qw + elems;
  unsigned short* vw = kw + elems;
  unsigned short* aw = vw + elems;

  dim3 gg(HH / 128, (BB * SS) / 128);   // 16 x 64
  dim3 bb(256);
  gemm_bias_k<false, 1><<<gg, bb, 0, stream>>>(x, Wq, bq, qw);
  gemm_bias_k<false, 1><<<gg, bb, 0, stream>>>(x, Wk, bk, kw);
  gemm_bias_k<false, 2><<<gg, bb, 0, stream>>>(x, Wv, bv, vw);  // V transposed
  flash_attn_k<<<dim3(SS / 128, BB * NHD), bb, 0, stream>>>(qw, kw, vw, aw);
  gemm_bias_k<true, 0><<<gg, bb, 0, stream>>>(aw, Wo, bo, d_out);
}
